// ClusterisedLinearNetworkSGD_47107201303342
// MI455X (gfx1250) — matrix operand order repaired and checked by reading, not compiled, not run on hardware
//
#include <hip/hip_runtime.h>

// ---------------------------------------------------------------------------
// ClusterisedLinearNetworkSGD for MI455X (gfx1250, wave32, WMMA).
//
// Pass 1 (prep_weights): fp32 -> f16 weight conversion with K-padding
//   W0[4,256,66] -> W0h[4,256,96], W1[4,256,256] -> W1h, W2[4,3,256] -> W2h[4,16,256]
//   Row-major f16 so a B-fragment (32x16, col n = Wt row n) is one contiguous
//   32B load per lane.  Also zeroes the expert bin counters.
// Pass 2 (point_prep): per-point NeRF encoding, argmin/softmax over KNN,
//   gathered W_lin heads (only 12 of 192 rows needed), builds inp[96] f16,
//   bins point index into per-expert compaction lists via atomicAdd.
// Pass 3 (mlp_expert_wmma): per (expert, 32-point tile) dense MLP
//   66->256->256->3 on v_wmma_f32_16x16x32_f16.  M=32 per wave (two 16-row
//   M-tiles) so each B fragment feeds 2 WMMAs -> half the L2 weight traffic
//   of M=16; A fragments register-cached across all N-tiles of a layer.
//
// Workspace (~105 MB): counts(4) | lists(4*N) | inp_h(N*96 f16) | W0h/W1h/W2h.
// ---------------------------------------------------------------------------

typedef __attribute__((ext_vector_type(16))) _Float16 v16h;
typedef __attribute__((ext_vector_type(8)))  _Float16 v8h;
typedef __attribute__((ext_vector_type(8)))  float    v8f;

#define KNN_E     4
#define ENC_DIM   63
#define KPAD0     96     // 66 padded to 3 K-chunks of 32
#define HID       256
#define NCLUST    64

// ---------------------------------------------------------------- pass 1 ---
__global__ void prep_weights(const float* __restrict__ W0,
                             const float* __restrict__ W1,
                             const float* __restrict__ W2,
                             _Float16* __restrict__ W0h,
                             _Float16* __restrict__ W1h,
                             _Float16* __restrict__ W2h,
                             int* __restrict__ counts) {
  int t = blockIdx.x * 256 + threadIdx.x;
  if (t < KNN_E) counts[t] = 0;
  const int n0 = KNN_E * HID * KPAD0;   // 98304
  const int n1 = KNN_E * HID * HID;     // 262144
  const int n2 = KNN_E * 16 * HID;      // 16384
  if (t < n0) {
    int c = t % KPAD0, r = (t / KPAD0) % HID, k = t / (KPAD0 * HID);
    W0h[t] = (_Float16)((c < 66) ? W0[((size_t)k * HID + r) * 66 + c] : 0.0f);
  } else if (t < n0 + n1) {
    int u = t - n0;
    W1h[u] = (_Float16)W1[u];
  } else if (t < n0 + n1 + n2) {
    int u = t - n0 - n1;
    int c = u % HID, r = (u / HID) % 16, k = u / (HID * 16);
    W2h[u] = (_Float16)((r < 3) ? W2[((size_t)k * 3 + r) * HID + c] : 0.0f);
  }
}

// ---------------------------------------------------------------- pass 2 ---
__global__ void point_prep(const float* __restrict__ X,
                           const int* __restrict__ cids,
                           const float* __restrict__ dist,
                           const float* __restrict__ W_lin,
                           const float* __restrict__ lambdas,
                           _Float16* __restrict__ inp_h,
                           int* __restrict__ counts,
                           int* __restrict__ list,
                           float* __restrict__ out, int N) {
  __shared__ float sW[NCLUST * 3 * ENC_DIM];   // 48384 B, L2 -> LDS once/block
  int tid = threadIdx.x;
  for (int j = tid; j < NCLUST * 3 * ENC_DIM; j += 256) sW[j] = W_lin[j];
  __syncthreads();

  int i = blockIdx.x * 256 + tid;
  if (i >= N) return;

  float x0 = X[3 * i], x1 = X[3 * i + 1], x2 = X[3 * i + 2];
  bool mask = (x0 == -1.0f) && (x1 == -1.0f) && (x2 == -1.0f);

  float enc[ENC_DIM];
  enc[0] = x0; enc[1] = x1; enc[2] = x2;
  float xv[3] = {x0, x1, x2};
  float freq = 1.0f;
#pragma unroll
  for (int f = 0; f < 10; ++f) {
#pragma unroll
    for (int j = 0; j < 3; ++j) {
      float s, c;
      __sincosf(freq * xv[j], &s, &c);
      enc[3 + f * 3 + j]  = s;   // sin block: enc[3..32]
      enc[33 + f * 3 + j] = c;   // cos block: enc[33..62]
    }
    freq *= 2.0f;
  }

  float d[KNN_E];
#pragma unroll
  for (int k = 0; k < KNN_E; ++k) d[k] = dist[(size_t)k * N + i];
  int nearest = 0; float dm = d[0];
#pragma unroll
  for (int k = 1; k < KNN_E; ++k)
    if (d[k] < dm) { dm = d[k]; nearest = k; }   // first-min, like jnp.argmin

  float lam = lambdas[nearest];
  float sc[KNN_E]; float smax = -1e30f;
#pragma unroll
  for (int k = 0; k < KNN_E; ++k) {
    sc[k] = lam * __expf(-lam * d[k]);
    smax = fmaxf(smax, sc[k]);
  }
  float wk[KNN_E]; float wsum = 0.0f;
#pragma unroll
  for (int k = 0; k < KNN_E; ++k) { wk[k] = __expf(sc[k] - smax); wsum += wk[k]; }
  float winv = 1.0f / wsum;

  // Only the 4 gathered cluster heads are ever needed (12 of 192 rows).
  float comb[3] = {0.0f, 0.0f, 0.0f};
#pragma unroll
  for (int k = 0; k < KNN_E; ++k) {
    int cid = cids[i * KNN_E + k];
    const float* wr = &sW[cid * 3 * ENC_DIM];
    float wkk = wk[k] * winv;
#pragma unroll
    for (int j = 0; j < 3; ++j) {
      float acc = 0.0f;
#pragma unroll
      for (int c = 0; c < ENC_DIM; ++c) acc += enc[c] * wr[j * ENC_DIM + c];
      comb[j] += wkk * acc;
    }
  }

  // inp[96] = [combined(3), enc(63), zeros(30)] as f16, 16B-vectorized store
  union { _Float16 h[KPAD0]; uint4 q[12]; } buf;
  buf.h[0] = (_Float16)comb[0]; buf.h[1] = (_Float16)comb[1]; buf.h[2] = (_Float16)comb[2];
#pragma unroll
  for (int c = 0; c < ENC_DIM; ++c) buf.h[3 + c] = (_Float16)enc[c];
#pragma unroll
  for (int c = 66; c < KPAD0; ++c) buf.h[c] = (_Float16)0.0f;
  uint4* dst = (uint4*)(inp_h + (size_t)i * KPAD0);
#pragma unroll
  for (int q = 0; q < 12; ++q) dst[q] = buf.q[q];

  if (mask) {
    out[3 * i] = 0.0f; out[3 * i + 1] = 0.0f; out[3 * i + 2] = 0.0f;
  } else {
    int pos = atomicAdd(&counts[nearest], 1);
    list[(size_t)nearest * N + pos] = i;   // order-independent result per point
  }
}

// ---------------------------------------------------------------- pass 3 ---
// A-fragment (16x32 f16): lane l -> row l&15,
//   K = {base + (l<16?0:16) .. +15} contiguous: two 16B LDS loads.
__device__ __forceinline__ v16h load_a(const _Float16* base, int stride,
                                       int lane, int c) {
  int row = lane & 15;
  int lo = c * 32 + ((lane & 16) ? 16 : 0);
  const _Float16* p = base + row * stride + lo;
  v8h l0 = *(const v8h*)(p);
  v8h h0 = *(const v8h*)(p + 8);
  v16h a;
#pragma unroll
  for (int i = 0; i < 8; ++i) { a[i] = l0[i]; a[i + 8] = h0[i]; }
  return a;
}

__device__ __forceinline__ v8f wmma_f16(const v16h& a, const v16h& b, const v8f& c) {
  return __builtin_amdgcn_wmma_f32_16x16x32_f16(false, a, false, b, (short)0,
                                                c, false, false);
}

// One hidden layer, M=32: D[32, 16*nTiles] = relu(A x W^T).
// A-frags (both M-tiles, all K-chunks) are register-cached; each B fragment
// (32x16 f16: lane l -> col l&15, 16 contiguous K at (l<16?0:16), one 32B
// load) is consumed by TWO WMMAs.
template <int CNUM>
__device__ __forceinline__ void run_layer(const v16h aF[2][8],
                                          const _Float16* __restrict__ Wb,
                                          int Kpad, int lane,
                                          _Float16* sHout, int nTiles) {
  int n  = lane & 15;
  int kb = (lane & 16) ? 16 : 0;
  int mb = (lane & 16) ? 8 : 0;
  for (int nt = 0; nt < nTiles; ++nt) {
    v8f acc0 = {0.f, 0.f, 0.f, 0.f, 0.f, 0.f, 0.f, 0.f};
    v8f acc1 = {0.f, 0.f, 0.f, 0.f, 0.f, 0.f, 0.f, 0.f};
#pragma unroll
    for (int c = 0; c < CNUM; ++c) {
      const _Float16* bp = Wb + (size_t)(nt * 16 + n) * Kpad + c * 32 + kb;
      v16h b = *(const v16h*)bp;                 // one 32B load, two WMMAs
      acc0 = wmma_f16(aF[0][c], b, acc0);
      acc1 = wmma_f16(aF[1][c], b, acc1);
    }
#pragma unroll
    for (int r = 0; r < 8; ++r) {                // C/D layout writeback + relu
      sHout[(mb + r) * HID + nt * 16 + n]        = (_Float16)fmaxf(acc0[r], 0.0f);
      sHout[(16 + mb + r) * HID + nt * 16 + n]   = (_Float16)fmaxf(acc1[r], 0.0f);
    }
  }
}

__global__ void __launch_bounds__(64)
mlp_expert_wmma(const _Float16* __restrict__ inp_h,
                const int* __restrict__ list,
                const int* __restrict__ counts,
                const _Float16* __restrict__ W0h,
                const _Float16* __restrict__ W1h,
                const _Float16* __restrict__ W2h,
                float* __restrict__ out, int N) {
  __shared__ __align__(16) _Float16 sIn[2][32 * KPAD0]; // 2 waves x 6 KB
  __shared__ __align__(16) _Float16 sH[2][32 * HID];    // 2 waves x 16 KB
  __shared__ int sIdx[2][32];

  const int lane = threadIdx.x & 31;
  const int w    = threadIdx.x >> 5;
  const int e    = blockIdx.y;                 // expert 0..3
  const int cnt  = counts[e];
  const int tile = blockIdx.x * 2 + w;
  const int start = tile * 32;
  const int rem   = cnt - start;               // up to 32 valid rows
  const bool active = rem > 0;                 // wave-uniform -> EXEC all-1s

  if (active) {                                // gather 32 compacted rows
    int ridx = start + lane; if (ridx > cnt - 1) ridx = cnt - 1;
    int p = list[(size_t)e * N + ridx];
    sIdx[w][lane] = p;
    const v8h* src = (const v8h*)(inp_h + (size_t)p * KPAD0);
    v8h* dstp = (v8h*)(&sIn[w][lane * KPAD0]);
#pragma unroll
    for (int q = 0; q < 6; ++q) dstp[q] = src[q];   // 6 x 16B = 48 of the 96 f16
  }
  __syncthreads();

  v16h aF[2][8];
  if (active) {                                // layer 0: [32,96]x[96,256]
#pragma unroll
    for (int mt = 0; mt < 2; ++mt)
#pragma unroll
      for (int c = 0; c < 3; ++c)
        aF[mt][c] = load_a(&sIn[w][mt * 16 * KPAD0], KPAD0, lane, c);
    run_layer<3>(aF, W0h + (size_t)e * HID * KPAD0, KPAD0, lane, &sH[w][0], 16);
  }
  __syncthreads();

  if (active) {                                // layer 1: [32,256]x[256,256]
#pragma unroll
    for (int mt = 0; mt < 2; ++mt)
#pragma unroll
      for (int c = 0; c < 8; ++c)
        aF[mt][c] = load_a(&sH[w][mt * 16 * HID], HID, lane, c);
    // frags are register-resident before sH is overwritten (DS in-order/wave)
    run_layer<8>(aF, W1h + (size_t)e * HID * HID, HID, lane, &sH[w][0], 16);
  }
  __syncthreads();

  if (active) {                                // layer 2: [32,256]x[256,16pad]
#pragma unroll
    for (int mt = 0; mt < 2; ++mt)
#pragma unroll
      for (int c = 0; c < 8; ++c)
        aF[mt][c] = load_a(&sH[w][mt * 16 * HID], HID, lane, c);
    const _Float16* Wb = W2h + (size_t)e * 16 * HID;
    int n  = lane & 15;
    int kb = (lane & 16) ? 16 : 0;
    int mb = (lane & 16) ? 8 : 0;
    v8f acc0 = {0.f, 0.f, 0.f, 0.f, 0.f, 0.f, 0.f, 0.f};
    v8f acc1 = {0.f, 0.f, 0.f, 0.f, 0.f, 0.f, 0.f, 0.f};
#pragma unroll
    for (int c = 0; c < 8; ++c) {
      const _Float16* bp = Wb + (size_t)n * HID + c * 32 + kb;
      v16h b = *(const v16h*)bp;
      acc0 = wmma_f16(aF[0][c], b, acc0);
      acc1 = wmma_f16(aF[1][c], b, acc1);
    }
    if (n < 3) {                               // only rgb columns are real
#pragma unroll
      for (int r = 0; r < 8; ++r) {
        int m0 = mb + r;
        if (m0 < rem)      out[(size_t)sIdx[w][m0]     * 3 + n] = tanhf(acc0[r]);
        int m1 = 16 + mb + r;
        if (m1 < rem)      out[(size_t)sIdx[w][m1]     * 3 + n] = tanhf(acc1[r]);
      }
    }
  }
}

// ------------------------------------------------------------------ launch -
extern "C" void kernel_launch(void* const* d_in, const int* in_sizes, int n_in,
                              void* d_out, int out_size, void* d_ws, size_t ws_size,
                              hipStream_t stream) {
  const float* X       = (const float*)d_in[0];
  const int*   cids    = (const int*)  d_in[1];
  const float* dist    = (const float*)d_in[2];
  const float* W_lin   = (const float*)d_in[3];
  const float* lambdas = (const float*)d_in[4];
  const float* W0      = (const float*)d_in[5];
  const float* W1      = (const float*)d_in[6];
  const float* W2      = (const float*)d_in[7];
  float* out = (float*)d_out;
  const int N = in_sizes[0] / 3;

  // workspace carve (~105 MB total for N=500K)
  auto al = [](size_t x) { return (x + 255) & ~(size_t)255; };
  char* ws = (char*)d_ws;
  size_t off = 0;
  int* counts = (int*)(ws + off);        off += 256;
  int* list   = (int*)(ws + off);        off += al((size_t)KNN_E * N * sizeof(int));
  _Float16* inp_h = (_Float16*)(ws + off); off += al((size_t)N * KPAD0 * 2);
  _Float16* W0h = (_Float16*)(ws + off); off += al((size_t)KNN_E * HID * KPAD0 * 2);
  _Float16* W1h = (_Float16*)(ws + off); off += al((size_t)KNN_E * HID * HID * 2);
  _Float16* W2h = (_Float16*)(ws + off); off += al((size_t)KNN_E * 16 * HID * 2);

  const int totW = KNN_E * HID * KPAD0 + KNN_E * HID * HID + KNN_E * 16 * HID;
  prep_weights<<<(totW + 255) / 256, 256, 0, stream>>>(W0, W1, W2, W0h, W1h, W2h, counts);
  point_prep<<<(N + 255) / 256, 256, 0, stream>>>(X, cids, dist, W_lin, lambdas,
                                                  inp_h, counts, list, out, N);
  const int tiles32 = (N + 31) / 32;
  dim3 gridC((tiles32 + 1) / 2, KNN_E);
  mlp_expert_wmma<<<gridC, 64, 0, stream>>>(inp_h, list, counts,
                                            W0h, W1h, W2h, out, N);
}
